// GraphEncoder_82214263980519
// MI455X (gfx1250) — matrix operand order repaired and checked by reading, not compiled, not run on hardware
//
#include <hip/hip_runtime.h>
#include <hip/hip_bf16.h>

// -------- problem constants (match reference) --------
#define HID   128
#define FIN   16
#define LAT   64
#define NLAY  3
#define NG    1024

typedef _Float16 v16h __attribute__((ext_vector_type(16)));
typedef _Float16 v4h  __attribute__((ext_vector_type(4)));
typedef float    v8f  __attribute__((ext_vector_type(8)));

union FragU { v16h v; uint4 u[2]; };
union H8U   { uint4 u; _Float16 h[8]; };

// ---------------------------------------------------------------
// 1) transpose + convert W_msg[l][k][n] (f32) -> Wt[l][n][k] (f16)
// ---------------------------------------------------------------
__global__ void cvt_w_kernel(const float* __restrict__ W_msg,
                             _Float16* __restrict__ Wt, int total) {
    int i = blockIdx.x * blockDim.x + threadIdx.x;
    if (i >= total) return;                 // total = NLAY*HID*HID
    int l = i >> 14;                        // HID*HID = 16384
    int r = i & 16383;
    int k = r >> 7;
    int n = r & 127;
    Wt[(l << 14) + (n << 7) + k] = (_Float16)W_msg[i];
}

// ---------------------------------------------------------------
// 2) zero graph-embedding accumulator (G*HID floats)
// ---------------------------------------------------------------
__global__ void zero_kernel(float* __restrict__ p, int total) {
    int i = blockIdx.x * blockDim.x + threadIdx.x;
    if (i < total) p[i] = 0.0f;
}

// ---------------------------------------------------------------
// 3) input layer: h16 = relu(x @ W_in + b_in) as f16; also zero agg
//    one block per node (128 threads = one feature each)
// ---------------------------------------------------------------
__global__ __launch_bounds__(HID)
void input_kernel(const float* __restrict__ x, const float* __restrict__ W_in,
                  const float* __restrict__ b_in, _Float16* __restrict__ h16,
                  float* __restrict__ agg, int Nn) {
    int node = blockIdx.x;
    int f = threadIdx.x;
    __shared__ float sx[FIN];
    if (f < FIN) sx[f] = x[node * FIN + f];
    __syncthreads();
    float acc = b_in[f];
#pragma unroll
    for (int k = 0; k < FIN; ++k) acc = fmaf(sx[k], W_in[k * HID + f], acc);
    acc = acc > 0.0f ? acc : 0.0f;
    h16[node * HID + f] = (_Float16)acc;
    agg[node * HID + f] = 0.0f;
}

// ---------------------------------------------------------------
// 4) WMMA GEMM: m16[N,128] = f16( h16[N,128] @ W[128,128] + b )
//    W given N-major f16 (Wt). Block: 256 thr = 8 waves; block tile
//    128 rows x 128 cols; wave w -> rows [w*16, w*16+16), all 8 col tiles.
//    acc starts at 0 (inline-0 C on first WMMA); bias added in epilogue.
// ---------------------------------------------------------------
__global__ __launch_bounds__(256)
void gemm_wmma_kernel(const _Float16* __restrict__ h16,
                      const _Float16* __restrict__ Wt,     // [128][128] n-major
                      const float* __restrict__ bias,      // [128]
                      _Float16* __restrict__ m16, int Nn) {
    __shared__ _Float16 sW[HID * HID];
    // stage W (32 KB) into LDS
    {
        const uint4* gw = reinterpret_cast<const uint4*>(Wt);
        uint4* sw = reinterpret_cast<uint4*>(sW);
#pragma unroll
        for (int i = threadIdx.x; i < (HID * HID) / 8; i += 256) sw[i] = gw[i];
    }
    __syncthreads();

    const int lane   = threadIdx.x & 31;
    const int wave   = threadIdx.x >> 5;
    const int ln     = lane & 15;
    const int hsel   = lane >> 4;          // 0: lanes 0-15, 1: lanes 16-31
    const int rowBase = blockIdx.x * 128 + wave * 16;

    // ---- load A fragments (16x32 f16, interleaved K layout) ----
    int arow = rowBase + ln;
    if (arow > Nn - 1) arow = Nn - 1;      // clamp: keep EXEC all-ones for WMMA
    const int akoff = hsel * 8;
    v16h afrag[4];
#pragma unroll
    for (int ks = 0; ks < 4; ++ks) {
        const int k0 = ks * 32;
        FragU t;
        t.u[0] = *reinterpret_cast<const uint4*>(h16 + arow * HID + k0 + akoff);
        t.u[1] = *reinterpret_cast<const uint4*>(h16 + arow * HID + k0 + akoff + 16);
        afrag[ks] = t.v;
    }

    const int bkbase = hsel * 8;           // B (32x16): interleaved K per lane-half
    const int srowBase = rowBase + hsel * 8;
    const bool fullTile = (rowBase + 16) <= Nn;   // wave-uniform store guard

#pragma unroll
    for (int ct = 0; ct < 8; ++ct) {
        const int col = ct * 16 + ln;
        const float bv = bias[col];
        v8f acc = {};                       // inline-0 C operand on first WMMA
#pragma unroll
        for (int ks = 0; ks < 4; ++ks) {
            const int k0 = ks * 32;
            FragU b;
            const uint4* bp =
                reinterpret_cast<const uint4*>(sW + col * HID + k0 + bkbase);
            b.u[0] = bp[0];
            b.u[1] = bp[2];
            acc = __builtin_amdgcn_wmma_f32_16x16x32_f16(
                false, afrag[ks], false, b.v, (short)0, acc, false, false);
        }
        // ---- store D: row = srowBase + r, col; bias added here ----
        if (fullTile) {
#pragma unroll
            for (int r = 0; r < 8; ++r)
                m16[(srowBase + r) * HID + col] = (_Float16)(acc[r] + bv);
        } else {
#pragma unroll
            for (int r = 0; r < 8; ++r) {
                const int row = srowBase + r;
                if (row < Nn) m16[row * HID + col] = (_Float16)(acc[r] + bv);
            }
        }
    }
}

// ---------------------------------------------------------------
// 5) edge scatter: agg[dst] += m16[src]
//    2 edges per wave: lanes 0-15 -> e0, lanes 16-31 -> e1;
//    each lane gathers 8 f16 (b128) and issues 8 f32 atomics.
// ---------------------------------------------------------------
__global__ __launch_bounds__(256)
void scatter_kernel(const int* __restrict__ src, const int* __restrict__ dst,
                    const _Float16* __restrict__ m16, float* __restrict__ agg,
                    int E) {
    const int gwave = (blockIdx.x * blockDim.x + threadIdx.x) >> 5;
    const int half  = (threadIdx.x >> 4) & 1;
    const int l16   = threadIdx.x & 15;
    const int e = gwave * 2 + half;
    if (e >= E) return;
    const int s = src[e];
    const int d = dst[e];
    H8U v;
    v.u = reinterpret_cast<const uint4*>(m16 + s * HID)[l16];  // 8 f16 per lane
    float* base = agg + d * HID + l16 * 8;
#pragma unroll
    for (int j = 0; j < 8; ++j) atomicAdd(base + j, (float)v.h[j]);
}

// ---------------------------------------------------------------
// 6) relu + convert to f16 for next layer; re-zero agg (float4 wide)
// ---------------------------------------------------------------
__global__ __launch_bounds__(256)
void relu_cvt_kernel(float* __restrict__ agg, _Float16* __restrict__ h16,
                     int total4) {
    int i = blockIdx.x * blockDim.x + threadIdx.x;
    if (i >= total4) return;
    float4 a = reinterpret_cast<float4*>(agg)[i];
    v4h h;
    h[0] = (_Float16)(a.x > 0.0f ? a.x : 0.0f);
    h[1] = (_Float16)(a.y > 0.0f ? a.y : 0.0f);
    h[2] = (_Float16)(a.z > 0.0f ? a.z : 0.0f);
    h[3] = (_Float16)(a.w > 0.0f ? a.w : 0.0f);
    reinterpret_cast<v4h*>(h16)[i] = h;
    reinterpret_cast<float4*>(agg)[i] = make_float4(0.f, 0.f, 0.f, 0.f);
}

// ---------------------------------------------------------------
// 7) final relu + graph sum-pool: ge[batch[n]] += relu(agg[n])
//    float4-wide: one batch lookup per 4 features
// ---------------------------------------------------------------
__global__ __launch_bounds__(256)
void relu_pool_kernel(const float* __restrict__ agg, const int* __restrict__ batch,
                      float* __restrict__ ge, int total4) {
    int i = blockIdx.x * blockDim.x + threadIdx.x;   // total4 = Nn*HID/4
    if (i >= total4) return;
    const int node = i >> 5;                         // 32 float4 per node
    const int f4 = i & 31;
    float4 a = reinterpret_cast<const float4*>(agg)[i];
    float* base = ge + batch[node] * HID + f4 * 4;
    atomicAdd(base + 0, a.x > 0.0f ? a.x : 0.0f);
    atomicAdd(base + 1, a.y > 0.0f ? a.y : 0.0f);
    atomicAdd(base + 2, a.z > 0.0f ? a.z : 0.0f);
    atomicAdd(base + 3, a.w > 0.0f ? a.w : 0.0f);
}

// ---------------------------------------------------------------
// 8) heads: z_mean = ge@W_mean+b_mean ; z_logvar = ge@W_logvar+b_logvar
// ---------------------------------------------------------------
__global__ __launch_bounds__(256)
void head_kernel(const float* __restrict__ ge,
                 const float* __restrict__ Wm, const float* __restrict__ bm,
                 const float* __restrict__ Wl, const float* __restrict__ bl,
                 float* __restrict__ out) {
    int i = blockIdx.x * blockDim.x + threadIdx.x;   // 2*NG*LAT total
    if (i >= 2 * NG * LAT) return;
    const int which = i >= NG * LAT;
    const int r = which ? (i - NG * LAT) : i;
    const int g = r / LAT;
    const int j = r % LAT;
    const float* W = which ? Wl : Wm;
    float acc = which ? bl[j] : bm[j];
    const float* row = ge + g * HID;
#pragma unroll 8
    for (int k = 0; k < HID; ++k) acc = fmaf(row[k], W[k * LAT + j], acc);
    out[i] = acc;
}

// ---------------------------------------------------------------
extern "C" void kernel_launch(void* const* d_in, const int* in_sizes, int n_in,
                              void* d_out, int out_size, void* d_ws, size_t ws_size,
                              hipStream_t stream) {
    const float* x        = (const float*)d_in[0];
    const int*   eidx     = (const int*)d_in[1];
    const int*   batch    = (const int*)d_in[2];
    const float* W_in     = (const float*)d_in[4];
    const float* b_in     = (const float*)d_in[5];
    const float* W_msg    = (const float*)d_in[6];
    const float* b_msg    = (const float*)d_in[7];
    const float* W_mean   = (const float*)d_in[8];
    const float* b_mean   = (const float*)d_in[9];
    const float* W_logvar = (const float*)d_in[10];
    const float* b_logvar = (const float*)d_in[11];

    const int Nn = in_sizes[0] / FIN;
    const int E  = in_sizes[1] / 2;
    const int* src = eidx;
    const int* dst = eidx + E;

    // workspace carve-up (256B aligned blocks)
    char* ws = (char*)d_ws;
    _Float16* Wt16 = (_Float16*)ws;                          // 3*16384*2 B
    size_t off = ((size_t)NLAY * HID * HID * 2 + 255) & ~(size_t)255;
    _Float16* h16 = (_Float16*)(ws + off);                   // N*128 f16
    off += ((size_t)Nn * HID * 2 + 255) & ~(size_t)255;
    _Float16* m16 = (_Float16*)(ws + off);                   // N*128 f16
    off += ((size_t)Nn * HID * 2 + 255) & ~(size_t)255;
    float* agg = (float*)(ws + off);                         // N*128 f32
    off += ((size_t)Nn * HID * 4 + 255) & ~(size_t)255;
    float* ge = (float*)(ws + off);                          // G*128 f32

    // 1) weight transpose/convert
    {
        int total = NLAY * HID * HID;
        cvt_w_kernel<<<(total + 255) / 256, 256, 0, stream>>>(W_msg, Wt16, total);
    }
    // 2) zero ge
    zero_kernel<<<(NG * HID + 255) / 256, 256, 0, stream>>>(ge, NG * HID);
    // 3) input layer (+ zero agg)
    input_kernel<<<Nn, HID, 0, stream>>>(x, W_in, b_in, h16, agg, Nn);

    const int gemmGrid = (Nn + 127) / 128;
    const int scatGrid = (E + 15) / 16;          // 16 edges per block (2/wave)
    const int reluGrid = (Nn * HID / 4 + 255) / 256;
    const int poolGrid = (Nn * HID / 4 + 255) / 256;

    for (int l = 0; l < NLAY; ++l) {
        gemm_wmma_kernel<<<gemmGrid, 256, 0, stream>>>(
            h16, Wt16 + (size_t)l * HID * HID, b_msg + l * HID, m16, Nn);
        scatter_kernel<<<scatGrid, 256, 0, stream>>>(src, dst, m16, agg, E);
        if (l < NLAY - 1)
            relu_cvt_kernel<<<reluGrid, 256, 0, stream>>>(agg, h16, Nn * HID / 4);
        else
            relu_pool_kernel<<<poolGrid, 256, 0, stream>>>(agg, batch, ge,
                                                           Nn * HID / 4);
    }
    // 8) heads
    head_kernel<<<(2 * NG * LAT + 255) / 256, 256, 0, stream>>>(
        ge, W_mean, b_mean, W_logvar, b_logvar, (float*)d_out);
}
